// TinyMHA_19739669693151
// MI455X (gfx1250) — matrix operand order repaired and checked by reading, not compiled, not run on hardware
//
#include <hip/hip_runtime.h>

// ---------------------------------------------------------------------------
// TinyMHA for MI455X (gfx1250): f16 WMMA GEMMs + flash attention, fp32 accum.
// Double-buffered LDS pipelines, templated epilogues, transpose-free layouts,
// permlane16 softmax max-reduction, row-sum via WMMA against a ones matrix.
// ---------------------------------------------------------------------------

typedef _Float16 v16h __attribute__((ext_vector_type(16)));
typedef _Float16 h8   __attribute__((ext_vector_type(8)));
typedef _Float16 h4   __attribute__((ext_vector_type(4)));
typedef float    v8f  __attribute__((ext_vector_type(8)));

#define SEQ 4096
#define DM  512
#define NH  8
#define DH  64

// Build a 16-half A/B fragment from two 16B-aligned 8-half chunks.
__device__ __forceinline__ v16h ld16(const _Float16* p0, const _Float16* p1) {
  h8 lo = *(const h8*)p0;
  h8 hi = *(const h8*)p1;
  return __builtin_shufflevector(lo, hi, 0,1,2,3,4,5,6,7,8,9,10,11,12,13,14,15);
}

__device__ __forceinline__ v8f wmma16(v16h a, v16h b, v8f c) {
  // D = A(16x32 f16) * B(32x16 f16) + C(16x16 f32)
  return __builtin_amdgcn_wmma_f32_16x16x32_f16(false, a, false, b, (short)0, c,
                                                false, false);
}

// xor-butterfly across each 16-lane row using v_permlane16_b32 (VALU, not DS).
// {sel_hi:sel_lo} nibble i = source lane (within row) for dest lane i.
__device__ __forceinline__ float pl16(float v, unsigned lo, unsigned hi) {
  unsigned u = __builtin_amdgcn_permlane16(__float_as_uint(v), __float_as_uint(v),
                                           lo, hi, false, false);
  return __uint_as_float(u);
}
__device__ __forceinline__ float rmax16(float v) {
  v = fmaxf(v, pl16(v, 0x67452301u, 0xEFCDAB89u));  // xor 1
  v = fmaxf(v, pl16(v, 0x54761032u, 0xDCFE98BAu));  // xor 2
  v = fmaxf(v, pl16(v, 0x32107654u, 0xBA98FEDCu));  // xor 4
  v = fmaxf(v, pl16(v, 0xFEDCBA98u, 0x76543210u));  // xor 8
  return v;
}

// ---------------------------------------------------------------------------
// fp32 -> f16 conversion (vectorized x4)
// ---------------------------------------------------------------------------
__global__ __launch_bounds__(256)
void cvt_f32_f16(const float* __restrict__ in, _Float16* __restrict__ out, int n4) {
  int i = blockIdx.x * blockDim.x + threadIdx.x;
  if (i < n4) {
    float4 f = ((const float4*)in)[i];
    h4 o = { (_Float16)f.x, (_Float16)f.y, (_Float16)f.z, (_Float16)f.w };
    ((h4*)out)[i] = o;
  }
}

// ---------------------------------------------------------------------------
// C[M=8192, E=512] = X[M,512](f16) * W[E,512]^T(f16) + bias(f32)
// MODE 0: Q  -> f16 [b,h,n,dh], scaled by 1/sqrt(64)
// MODE 1: K  -> f16 [b,h,n,dh]
// MODE 2: Vt -> f16 [b,h,dh,n]   (LDS-transposed epilogue, 64B stores)
// MODE 3: out-> f32 [b,n,e]
// Block: 256 thr = 8 waves; tile 128(M) x 128(E); wave = 32x64 (2x4 WMMA tiles)
// K-loop: step 32, double-buffered LDS (one barrier/step), ping-pong pointers.
// ---------------------------------------------------------------------------
template <int MODE>
__global__ __launch_bounds__(256)
void gemm_xwT(const _Float16* __restrict__ X, const _Float16* __restrict__ W,
              const float* __restrict__ bias, void* __restrict__ dst) {
  // two buffers of (X:128x40 + W:128x40) halves
  __shared__ __align__(16) _Float16 smem[2 * 10240];

  const int t    = threadIdx.x;
  const int wave = t >> 5;
  const int lane = t & 31;
  const int l16  = lane & 15;
  const int g    = lane >> 4;
  const int m0   = blockIdx.x * 128;
  const int e0   = blockIdx.y * 128;
  const int mw   = (wave >> 1) * 32;  // wave M offset
  const int ew   = (wave & 1) * 64;   // wave E offset
  const int srow = t >> 1, sch = (t & 1) * 16;  // staging: 16 halves/thread/tile

  const _Float16* Xs = X + (size_t)(m0 + srow) * DM + sch;
  const _Float16* Ws = W + (size_t)(e0 + srow) * DM + sch;

  _Float16* bufA = smem;           // read buffer this iteration
  _Float16* bufB = smem + 10240;   // write buffer this iteration

  // prologue: stage k0 = 0 into bufA
  uint4 xr0 = *(const uint4*)(Xs);
  uint4 xr1 = *(const uint4*)(Xs + 8);
  uint4 wr0 = *(const uint4*)(Ws);
  uint4 wr1 = *(const uint4*)(Ws + 8);
  *(uint4*)&bufA[srow * 40 + sch]            = xr0;
  *(uint4*)&bufA[srow * 40 + sch + 8]        = xr1;
  *(uint4*)&bufA[5120 + srow * 40 + sch]     = wr0;
  *(uint4*)&bufA[5120 + srow * 40 + sch + 8] = wr1;
  __syncthreads();

  v8f acc[2][4] = { { {}, {}, {}, {} }, { {}, {}, {}, {} } };

  for (int k0 = 0; k0 < DM; k0 += 32) {
    const bool nxt = (k0 + 32 < DM);
    if (nxt) {  // issue next-tile global loads early (latency hiding)
      xr0 = *(const uint4*)(Xs + k0 + 32);
      xr1 = *(const uint4*)(Xs + k0 + 40);
      wr0 = *(const uint4*)(Ws + k0 + 32);
      wr1 = *(const uint4*)(Ws + k0 + 40);
      if (k0 + 64 < DM) __builtin_prefetch(Xs + k0 + 64, 0, 1);
    }
    const _Float16* xb = bufA;
    const _Float16* wb = bufA + 5120;

    v16h a[2], b[4];
#pragma unroll
    for (int mt = 0; mt < 2; ++mt) {
      const int mr = mw + mt * 16 + l16;  // A: lane row m, k=16g+e (contiguous)
      a[mt] = ld16(&xb[mr * 40 + g * 16], &xb[mr * 40 + g * 16 + 8]);
    }
#pragma unroll
    for (int nt = 0; nt < 4; ++nt) {
      const int er = ew + nt * 16 + l16;  // B: lane col n, k=16g+e (contiguous)
      b[nt] = ld16(&wb[er * 40 + g * 16], &wb[er * 40 + g * 16 + 8]);
    }
#pragma unroll
    for (int mt = 0; mt < 2; ++mt)
#pragma unroll
      for (int nt = 0; nt < 4; ++nt) acc[mt][nt] = wmma16(a[mt], b[nt], acc[mt][nt]);

    if (nxt) {
      *(uint4*)&bufB[srow * 40 + sch]            = xr0;
      *(uint4*)&bufB[srow * 40 + sch + 8]        = xr1;
      *(uint4*)&bufB[5120 + srow * 40 + sch]     = wr0;
      *(uint4*)&bufB[5120 + srow * 40 + sch + 8] = wr1;
      __syncthreads();
      _Float16* tmp = bufA; bufA = bufB; bufB = tmp;
    }
  }

  // ------------------------- epilogue (templated) --------------------------
  if (MODE == 2) {
    // Transpose 64(e) x 32(token) wave tile through LDS, then 64B row stores.
    __syncthreads();  // done with GEMM buffers; reuse smem
    _Float16* tb = smem + wave * 2560;  // 64 rows x 32 (pad->40)
#pragma unroll
    for (int nt = 0; nt < 4; ++nt) {
      const float bv = bias[e0 + ew + nt * 16 + l16];
#pragma unroll
      for (int mt = 0; mt < 2; ++mt)
#pragma unroll
        for (int r = 0; r < 8; ++r)
          tb[(nt * 16 + l16) * 40 + mt * 16 + r + 8 * g] =
              (_Float16)(acc[mt][nt][r] + bv);
    }
    const int token0 = m0 + mw;
    const int b_ = token0 >> 12, n0 = token0 & (SEQ - 1);
#pragma unroll
    for (int rr = 0; rr < 2; ++rr) {
      const int row = lane + rr * 32;  // e_rel 0..63
      const int e = e0 + ew + row;
      _Float16* dp = (_Float16*)dst +
                     ((size_t)(b_ * NH + (e >> 6)) * DH + (e & (DH - 1))) * SEQ + n0;
      const _Float16* sp = &tb[row * 40];
#pragma unroll
      for (int c = 0; c < 4; ++c) *(uint4*)(dp + c * 8) = *(const uint4*)(sp + c * 8);
    }
  } else {
#pragma unroll
    for (int nt = 0; nt < 4; ++nt) {
      const int e = e0 + ew + nt * 16 + l16;
      const float bv = bias[e];
      const int hh = e >> 6, dd = e & (DH - 1);
#pragma unroll
      for (int mt = 0; mt < 2; ++mt) {
#pragma unroll
        for (int r = 0; r < 8; ++r) {
          const int token = m0 + mw + mt * 16 + r + 8 * g;
          const float v = acc[mt][nt][r] + bv;
          if (MODE == 0) {
            const int b_ = token >> 12, n_ = token & (SEQ - 1);
            ((_Float16*)dst)[((size_t)(b_ * NH + hh) * SEQ + n_) * DH + dd] =
                (_Float16)(v * 0.125f);
          } else if (MODE == 1) {
            const int b_ = token >> 12, n_ = token & (SEQ - 1);
            ((_Float16*)dst)[((size_t)(b_ * NH + hh) * SEQ + n_) * DH + dd] =
                (_Float16)v;
          } else {  // MODE == 3
            ((float*)dst)[(size_t)token * DM + e] = v;
          }
        }
      }
    }
  }
}

// ---------------------------------------------------------------------------
// Flash attention: block = (qblock of 128, head, batch); 8 waves x 16 q-rows.
// Q pre-scaled by 1/sqrt(dh). K: [b,h,N,dh] f16; Vt: [b,h,dh,N] f16.
// Key-block 32, double-buffered K/Vt staging (one barrier per key-block).
// Row max: permlane16 butterfly (VALU). Row sum: WMMA against ones matrix.
// ---------------------------------------------------------------------------
__global__ __launch_bounds__(256)
void flash_attn(const _Float16* __restrict__ Q, const _Float16* __restrict__ K,
                const _Float16* __restrict__ Vt, _Float16* __restrict__ Y) {
  __shared__ __align__(16) _Float16 ks[2][32 * 72];      // 32 keys x 64 (pad->72)
  __shared__ __align__(16) _Float16 vs[2][64 * 40];      // 64 d x 32 keys (pad->40)
  __shared__ __align__(16) _Float16 ps[8 * 16 * 40];     // per-wave P tile 16x32

  const int t    = threadIdx.x;
  const int wave = t >> 5;
  const int lane = t & 31;
  const int l16  = lane & 15;
  const int g    = lane >> 4;
  const int h    = blockIdx.y;
  const int b    = blockIdx.z;
  const int bh   = b * NH + h;
  const int q0   = blockIdx.x * 128 + wave * 16;

  const _Float16* Qbh = Q + (size_t)bh * SEQ * DH;
  const _Float16* Kbh = K + (size_t)bh * SEQ * DH;
  const _Float16* Vbh = Vt + (size_t)bh * DH * SEQ;

  // Resident Q A-fragments (16 rows x 64 dh = 2 fragments)
  v16h qa[2];
  {
    const _Float16* qrow = Qbh + (size_t)(q0 + l16) * DH;
#pragma unroll
    for (int j = 0; j < 2; ++j)
      qa[j] = ld16(qrow + j * 32 + g * 16, qrow + j * 32 + g * 16 + 8);
  }

  v16h ones;
#pragma unroll
  for (int e = 0; e < 16; ++e) ones[e] = (_Float16)1.0f;

  v8f o[4] = { {}, {}, {}, {} };
  float mst[8], lst[8];
#pragma unroll
  for (int r = 0; r < 8; ++r) { mst[r] = -INFINITY; lst[r] = 0.f; }
  _Float16* pw = &ps[wave * 16 * 40];

  const int krow = t >> 3, kch = (t & 7) * 8;  // K stage: 8 halves/thread
  const int vrow = t >> 2, vch = (t & 3) * 8;  // Vt stage: 8 halves/thread
  const _Float16* Kst = Kbh + (size_t)krow * DH + kch;
  const _Float16* Vst = Vbh + (size_t)vrow * SEQ + vch;

  _Float16* kA = ks[0]; _Float16* kB = ks[1];
  _Float16* vA = vs[0]; _Float16* vB = vs[1];

  // prologue: stage key-block 0 into buffer A
  uint4 kreg = *(const uint4*)(Kst);
  uint4 vreg = *(const uint4*)(Vst);
  *(uint4*)&kA[krow * 72 + kch] = kreg;
  *(uint4*)&vA[vrow * 40 + vch] = vreg;
  __syncthreads();

  for (int kb = 0; kb < SEQ; kb += 32) {
    const bool nxt = (kb + 32 < SEQ);
    if (nxt) {  // issue next-tile global loads early
      kreg = *(const uint4*)(Kst + (size_t)(kb + 32) * DH);
      vreg = *(const uint4*)(Vst + kb + 32);
      if (kb + 64 < SEQ) {
        __builtin_prefetch(Kst + (size_t)(kb + 64) * DH, 0, 1);
        __builtin_prefetch(Vst + kb + 64, 0, 1);
      }
    }

    // All 4 K B-fragments up front, then S WMMAs (lets dscnt stagger)
    v16h kfr[2][2];
#pragma unroll
    for (int c = 0; c < 2; ++c) {
      const int kr = (c * 16 + l16) * 72;
      kfr[c][0] = ld16(&kA[kr + g * 16], &kA[kr + g * 16 + 8]);            // d 0..31
      kfr[c][1] = ld16(&kA[kr + 32 + g * 16], &kA[kr + 32 + g * 16 + 8]);  // d 32..63
    }
    v8f s[2];
#pragma unroll
    for (int c = 0; c < 2; ++c) {
      v8f z = {};
      z = wmma16(qa[0], kfr[c][0], z);
      z = wmma16(qa[1], kfr[c][1], z);
      s[c] = z;
    }

    // Issue V B-fragment loads now; their DS latency hides under softmax VALU
    v16h vfr[4];
#pragma unroll
    for (int tt = 0; tt < 4; ++tt) {
      const int vr = (tt * 16 + l16) * 40;  // B: lane = d column, k = key
      vfr[tt] = ld16(&vA[vr + g * 16], &vA[vr + g * 16 + 8]);
    }

    // Online softmax per row (m = r + 8g); max via permlane16 butterfly
    float corr8[8];
#pragma unroll
    for (int r = 0; r < 8; ++r) {
      const float rv = rmax16(fmaxf(s[0][r], s[1][r]));
      const float mo = mst[r];
      const float mn = fmaxf(mo, rv);
      corr8[r] = __expf(mo - mn);
      const float p0 = __expf(s[0][r] - mn);
      const float p1 = __expf(s[1][r] - mn);
      mst[r] = mn;
#pragma unroll
      for (int tt = 0; tt < 4; ++tt) o[tt][r] *= corr8[r];
      const int m = r + 8 * g;
      pw[m * 40 + l16]      = (_Float16)p0;
      pw[m * 40 + 16 + l16] = (_Float16)p1;
    }

    // P (C-layout) -> A-fragment via per-wave LDS
    v16h pa = ld16(&pw[l16 * 40 + g * 16], &pw[l16 * 40 + g * 16 + 8]);

    // Row sums of P via WMMA against ones (replaces 32 ds_bpermute)
    v8f sums = {};
    sums = wmma16(pa, ones, sums);
#pragma unroll
    for (int r = 0; r < 8; ++r) lst[r] = lst[r] * corr8[r] + sums[r];

    // O += P * V
#pragma unroll
    for (int tt = 0; tt < 4; ++tt) o[tt] = wmma16(pa, vfr[tt], o[tt]);

    if (nxt) {  // stage next key-block into the other buffer
      *(uint4*)&kB[krow * 72 + kch] = kreg;
      *(uint4*)&vB[vrow * 40 + vch] = vreg;
      __syncthreads();
      _Float16* tp;
      tp = kA; kA = kB; kB = tp;
      tp = vA; vA = vB; vB = tp;
    }
  }

  // Normalize and store y[b, q, h*64 + d] as f16 (32B contiguous / 16 lanes)
#pragma unroll
  for (int r = 0; r < 8; ++r) {
    const float il = 1.0f / lst[r];
    const int qr = q0 + r + 8 * g;
    _Float16* dstp = Y + ((size_t)b * SEQ + qr) * DM + h * DH;
#pragma unroll
    for (int tt = 0; tt < 4; ++tt)
      dstp[tt * 16 + l16] = (_Float16)(o[tt][r] * il);
  }
}

// ---------------------------------------------------------------------------
// Host launcher
// ---------------------------------------------------------------------------
extern "C" void kernel_launch(void* const* d_in, const int* in_sizes, int n_in,
                              void* d_out, int out_size, void* d_ws, size_t ws_size,
                              hipStream_t stream) {
  const float* x  = (const float*)d_in[0];
  const float* wq = (const float*)d_in[1];
  const float* bq = (const float*)d_in[2];
  const float* wk = (const float*)d_in[3];
  const float* bk = (const float*)d_in[4];
  const float* wv = (const float*)d_in[5];
  const float* bv = (const float*)d_in[6];
  const float* wo = (const float*)d_in[7];
  const float* bo = (const float*)d_in[8];

  const size_t M  = 2 * (size_t)SEQ;   // 8192 tokens
  const size_t xN = M * DM;            // 4,194,304
  const size_t wN = (size_t)DM * DM;   // 262,144

  char* ws = (char*)d_ws;
  size_t off = 0;
  auto carve = [&](size_t bytes) { void* p = ws + off; off += bytes; return p; };
  _Float16* xh  = (_Float16*)carve(xN * 2);
  _Float16* wqh = (_Float16*)carve(wN * 2);
  _Float16* wkh = (_Float16*)carve(wN * 2);
  _Float16* wvh = (_Float16*)carve(wN * 2);
  _Float16* woh = (_Float16*)carve(wN * 2);
  _Float16* qh  = (_Float16*)carve(xN * 2);  // [b,h,n,dh], pre-scaled
  _Float16* kh  = (_Float16*)carve(xN * 2);  // [b,h,n,dh]
  _Float16* vth = (_Float16*)carve(xN * 2);  // [b,h,dh,n]
  _Float16* yh  = (_Float16*)carve(xN * 2);  // [b,n,d]

  // fp32 -> f16
  cvt_f32_f16<<<(int)(xN / 4 + 255) / 256, 256, 0, stream>>>(x, xh, (int)(xN / 4));
  cvt_f32_f16<<<(int)(wN / 4 + 255) / 256, 256, 0, stream>>>(wq, wqh, (int)(wN / 4));
  cvt_f32_f16<<<(int)(wN / 4 + 255) / 256, 256, 0, stream>>>(wk, wkh, (int)(wN / 4));
  cvt_f32_f16<<<(int)(wN / 4 + 255) / 256, 256, 0, stream>>>(wv, wvh, (int)(wN / 4));
  cvt_f32_f16<<<(int)(wN / 4 + 255) / 256, 256, 0, stream>>>(wo, woh, (int)(wN / 4));

  // QKV projections (tile 128x128)
  dim3 gg((unsigned)(M / 128), DM / 128);
  gemm_xwT<0><<<gg, 256, 0, stream>>>(xh, wqh, bq, qh);
  gemm_xwT<1><<<gg, 256, 0, stream>>>(xh, wkh, bk, kh);
  gemm_xwT<2><<<gg, 256, 0, stream>>>(xh, wvh, bv, vth);

  // Flash attention
  dim3 ga(SEQ / 128, NH, 2);
  flash_attn<<<ga, 256, 0, stream>>>(qh, kh, vth, yh);

  // Output projection -> fp32 d_out
  gemm_xwT<3><<<gg, 256, 0, stream>>>(yh, woh, bo, d_out);
}
